// AntisymMP_18528488915346
// MI455X (gfx1250) — compile-verified, matrix operand order checked by reading, not run on hardware
//
#include <hip/hip_runtime.h>
#include <hip/hip_bf16.h>
#include <math.h>

// CDNA5 / gfx1250: wave32, WMMA 16x16x32 bf16 -> f32 accumulators.
typedef __attribute__((ext_vector_type(16))) __bf16 v16bf;
typedef __attribute__((ext_vector_type(8)))  __bf16 v8bf;
typedef __attribute__((ext_vector_type(8)))  float  v8f;

// 128-bit chunk type matching the async-to-LDS builtin's parameter type.
typedef int v4i_t __attribute__((vector_size(16)));
typedef __attribute__((address_space(1))) v4i_t glob_v4i;
typedef __attribute__((address_space(3))) v4i_t lds_v4i;

// Build a 16-element bf16 A/B fragment from two 16-byte LDS chunks.
// Per ISA (CDNA5 ISA): lanes 0-15 hold K={0..7,16..23},
// lanes 16-31 hold K={8..15,24..31} for 16-bit 16x32 operands.
static __device__ __forceinline__ v16bf load_frag(const __bf16* p0, const __bf16* p1) {
  v8bf a = *(const v8bf*)p0;
  v8bf b = *(const v8bf*)p1;
  v16bf r;
#pragma unroll
  for (int i = 0; i < 8; ++i) { r[i] = a[i]; r[i + 8] = b[i]; }
  return r;
}

static __device__ __forceinline__ v8f vzero8() {
  v8f z = {0.f, 0.f, 0.f, 0.f, 0.f, 0.f, 0.f, 0.f};
  return z;
}

// 16-byte memory -> LDS copy. On gfx1250 use the async-to-LDS DMA path
// (GLOBAL_LOAD_ASYNC_TO_LDS_B128, tracked by ASYNCcnt) so the copy bypasses
// VGPRs and overlaps the VALU-heavy gather staging; fall back to a plain
// load/store if the builtin is absent on this toolchain.
static __device__ __forceinline__ void copy16_to_lds(const __bf16* gsrc, __bf16* ldst) {
#if __has_builtin(__builtin_amdgcn_global_load_async_to_lds_b128)
  __builtin_amdgcn_global_load_async_to_lds_b128(
      (glob_v4i*)gsrc, (lds_v4i*)ldst, 0, 0);
#else
  *(v8bf*)ldst = *(const v8bf*)gsrc;
#endif
}

static __device__ __forceinline__ void wait_async_copies() {
#if __has_builtin(__builtin_amdgcn_global_load_async_to_lds_b128)
#if __has_builtin(__builtin_amdgcn_s_wait_asynccnt)
  __builtin_amdgcn_s_wait_asynccnt(0);
#else
  asm volatile("s_wait_asynccnt 0x0" ::: "memory");
#endif
#endif
}

// Stage one 256x32 pre-transposed bf16 weight slice (16 KB) into sW.
// Layout in memory is already [kstep][c][kk] so each thread copies a
// contiguous 64 bytes as 4 async b128 transfers.
static __device__ __forceinline__ void stage_weights(const __bf16* Wt, int ks,
                                                     __bf16* sW, int tid) {
  const __bf16* g = Wt + ((size_t)ks << 13) + (tid << 5);
  __bf16* l = sW + (tid << 5);
#pragma unroll
  for (int q = 0; q < 4; ++q) copy16_to_lds(g + q * 8, l + q * 8);
}

// ---------------------------------------------------------------------------
// One-time weight conversion: f32 row-major W[k][c] -> bf16 staged layout
// [kstep][c][kk] (kstep = k/32, kk = k%32), 56 k-steps total:
//   W_e1: ks 0..23   W_e2: ks 24..31   W_n1: ks 32..47   W_n2: ks 48..55
// ---------------------------------------------------------------------------
__global__ __launch_bounds__(256) void convert_weights_kernel(
    const float* __restrict__ We1, const float* __restrict__ We2,
    const float* __restrict__ Wn1, const float* __restrict__ Wn2,
    __bf16* __restrict__ out)
{
  int i = blockIdx.x * 256 + threadIdx.x;
  const int total = 56 * 8192;
  if (i >= total) return;
  int seg = i >> 13;            // global k-step 0..55
  int rem = i & 8191;
  int c = rem >> 5, kk = rem & 31;
  const float* W; int ksl;
  if (seg < 24)      { W = We1; ksl = seg; }
  else if (seg < 32) { W = We2; ksl = seg - 24; }
  else if (seg < 48) { W = Wn1; ksl = seg - 32; }
  else               { W = Wn2; ksl = seg - 48; }
  out[i] = (__bf16)W[(ksl * 32 + kk) * 256 + c];
}

// ---------------------------------------------------------------------------
// Edge kernel: rows of X (== edge index) are [ha,hb,es] / [hb,ha,es] pairs.
// Block: 256 threads = 8 waves, computes 64 rows x 256 cols.
// GEMM1: K=768 (h[a] | h[b] | 0.5*(e[2k]+e[2k+1])), silu -> LDS bf16.
// GEMM2: K=256. Epilogue: mf = Y[2k]-Y[2k+1] (b_e2 cancels), writes e2 and
// atomically scatters +/-mf into agg.
// ---------------------------------------------------------------------------
__global__ __launch_bounds__(256) void edge_mlp_kernel(
    const float* __restrict__ h, const float* __restrict__ e,
    const __bf16* __restrict__ Wt1, const float* __restrict__ b1,
    const __bf16* __restrict__ Wt2,
    const int* __restrict__ src, const int* __restrict__ dst,
    float* __restrict__ e2, float* __restrict__ agg, int E)
{
  __shared__ __align__(16) __bf16 sA[64 * 32];    //  4 KB staged X tile
  __shared__ __align__(16) __bf16 sW[256 * 32];   // 16 KB staged Wt slice
  __shared__ __align__(16) __bf16 sI[64 * 256];   // 32 KB silu intermediate

  const int tid    = threadIdx.x;
  const int lane   = tid & 31;
  const int wave   = tid >> 5;
  const int waveR  = wave >> 1;        // 0..3 : row tile
  const int waveC  = wave & 1;         // 0..1 : col half
  const int lanelo = lane & 15;
  const int hi     = (lane >> 4) & 1;
  const int R0     = blockIdx.x * 64;  // first X row == first edge index

  v8f acc[8];
#pragma unroll
  for (int t = 0; t < 8; ++t) acc[t] = vzero8();

  // -------- GEMM1: K = 768, 24 k-steps of 32 --------
  for (int ks = 0; ks < 24; ++ks) {
    const int k0 = ks * 32;
    __syncthreads();                    // prior k-step fully consumed
    stage_weights(Wt1, ks, sW, tid);    // DMA in flight while we gather X
    // Stage X tile (gather + f32->bf16), 8 elements per thread.
#pragma unroll
    for (int j = 0; j < 8; ++j) {
      int idx = j * 256 + tid;
      int r = idx >> 5, kk = idx & 31;
      int er = R0 + r;                 // edge / X row
      float v = 0.0f;
      if (er < E) {
        int pe = er & ~1;              // forward-edge index of the pair
        int kc = k0 + kk;
        if (kc < 512) {
          int a = src[pe], bn = dst[pe];
          int dir = er & 1;
          int node = (kc < 256) ? (dir ? bn : a) : (dir ? a : bn);
          v = h[node * 256 + (kc & 255)];
        } else {
          int c = kc - 512;
          v = 0.5f * (e[pe * 256 + c] + e[(pe + 1) * 256 + c]);
        }
      }
      sA[r * 32 + kk] = (__bf16)v;
    }
    wait_async_copies();
    __syncthreads();

    int arow = waveR * 16 + lanelo;
    v16bf af = load_frag(&sA[arow * 32 + hi * 8], &sA[arow * 32 + hi * 8 + 16]);
#pragma unroll
    for (int t = 0; t < 8; ++t) {
      int ct = waveC * 128 + t * 16 + lanelo;
      v16bf bf = load_frag(&sW[ct * 32 + hi * 8], &sW[ct * 32 + hi * 8 + 16]);
      acc[t] = __builtin_amdgcn_wmma_f32_16x16x32_bf16(
          false, af, false, bf, (short)0, acc[t], false, false);
    }
  }

  // bias + silu -> sI (bf16). C layout: VGPR j = row j (lo lanes) / 8+j (hi).
  __syncthreads();
#pragma unroll
  for (int t = 0; t < 8; ++t) {
    int c = waveC * 128 + t * 16 + lanelo;
    float bb = b1[c];
#pragma unroll
    for (int j = 0; j < 8; ++j) {
      int row = waveR * 16 + hi * 8 + j;
      float x = acc[t][j] + bb;
      float s = x / (1.0f + __expf(-x));
      sI[row * 256 + c] = (__bf16)s;
    }
    acc[t] = vzero8();
  }

  // -------- GEMM2: K = 256, 8 k-steps --------
  for (int ks = 0; ks < 8; ++ks) {
    const int k0 = ks * 32;
    __syncthreads();
    stage_weights(Wt2, ks, sW, tid);
    wait_async_copies();
    __syncthreads();

    int arow = waveR * 16 + lanelo;
    v16bf af = load_frag(&sI[arow * 256 + k0 + hi * 8],
                         &sI[arow * 256 + k0 + hi * 8 + 16]);
#pragma unroll
    for (int t = 0; t < 8; ++t) {
      int ct = waveC * 128 + t * 16 + lanelo;
      v16bf bf = load_frag(&sW[ct * 32 + hi * 8], &sW[ct * 32 + hi * 8 + 16]);
      acc[t] = __builtin_amdgcn_wmma_f32_16x16x32_bf16(
          false, af, false, bf, (short)0, acc[t], false, false);
    }
  }

  // -------- Epilogue: mf = Y[2k] - Y[2k+1]  (b_e2 cancels) --------
  const int Rw = R0 + waveR * 16;
#pragma unroll
  for (int t = 0; t < 8; ++t) {
    int c = waveC * 128 + t * 16 + lanelo;
    int rbase = Rw + hi * 8;
#pragma unroll
    for (int i = 0; i < 4; ++i) {
      int eA = rbase + 2 * i;
      if (eA < E) {
        int eB = eA + 1;
        float mf = acc[t][2 * i] - acc[t][2 * i + 1];
        e2[eA * 256 + c] = e[eA * 256 + c] + mf;
        e2[eB * 256 + c] = e[eB * 256 + c] - mf;
        atomicAdd(agg + dst[eA] * 256 + c,  mf);
        atomicAdd(agg + dst[eB] * 256 + c, -mf);
      }
    }
  }
}

// ---------------------------------------------------------------------------
// Node kernel: X = [h | agg] (K=512), out_h = h + silu-MLP(X), written
// in place over the agg region (row-local, each block owns its rows).
// ---------------------------------------------------------------------------
__global__ __launch_bounds__(256) void node_mlp_kernel(
    const float* __restrict__ h,
    const __bf16* __restrict__ Wt1, const float* __restrict__ b1,
    const __bf16* __restrict__ Wt2, const float* __restrict__ b2,
    float* __restrict__ agg_out, int Nn)
{
  __shared__ __align__(16) __bf16 sA[64 * 32];
  __shared__ __align__(16) __bf16 sW[256 * 32];
  __shared__ __align__(16) __bf16 sI[64 * 256];

  const int tid    = threadIdx.x;
  const int lane   = tid & 31;
  const int wave   = tid >> 5;
  const int waveR  = wave >> 1;
  const int waveC  = wave & 1;
  const int lanelo = lane & 15;
  const int hi     = (lane >> 4) & 1;
  const int R0     = blockIdx.x * 64;

  v8f acc[8];
#pragma unroll
  for (int t = 0; t < 8; ++t) acc[t] = vzero8();

  // -------- GEMM1: K = 512, 16 k-steps --------
  for (int ks = 0; ks < 16; ++ks) {
    const int k0 = ks * 32;
    __syncthreads();
    stage_weights(Wt1, ks, sW, tid);
#pragma unroll
    for (int j = 0; j < 8; ++j) {
      int idx = j * 256 + tid;
      int r = idx >> 5, kk = idx & 31;
      int rg = R0 + r;
      float v = 0.0f;
      if (rg < Nn) {
        int kc = k0 + kk;
        v = (kc < 256) ? h[rg * 256 + kc] : agg_out[rg * 256 + (kc - 256)];
      }
      sA[r * 32 + kk] = (__bf16)v;
    }
    wait_async_copies();
    __syncthreads();

    int arow = waveR * 16 + lanelo;
    v16bf af = load_frag(&sA[arow * 32 + hi * 8], &sA[arow * 32 + hi * 8 + 16]);
#pragma unroll
    for (int t = 0; t < 8; ++t) {
      int ct = waveC * 128 + t * 16 + lanelo;
      v16bf bf = load_frag(&sW[ct * 32 + hi * 8], &sW[ct * 32 + hi * 8 + 16]);
      acc[t] = __builtin_amdgcn_wmma_f32_16x16x32_bf16(
          false, af, false, bf, (short)0, acc[t], false, false);
    }
  }

  __syncthreads();
#pragma unroll
  for (int t = 0; t < 8; ++t) {
    int c = waveC * 128 + t * 16 + lanelo;
    float bb = b1[c];
#pragma unroll
    for (int j = 0; j < 8; ++j) {
      int row = waveR * 16 + hi * 8 + j;
      float x = acc[t][j] + bb;
      float s = x / (1.0f + __expf(-x));
      sI[row * 256 + c] = (__bf16)s;
    }
    acc[t] = vzero8();
  }

  // -------- GEMM2: K = 256 --------
  for (int ks = 0; ks < 8; ++ks) {
    const int k0 = ks * 32;
    __syncthreads();
    stage_weights(Wt2, ks, sW, tid);
    wait_async_copies();
    __syncthreads();

    int arow = waveR * 16 + lanelo;
    v16bf af = load_frag(&sI[arow * 256 + k0 + hi * 8],
                         &sI[arow * 256 + k0 + hi * 8 + 16]);
#pragma unroll
    for (int t = 0; t < 8; ++t) {
      int ct = waveC * 128 + t * 16 + lanelo;
      v16bf bf = load_frag(&sW[ct * 32 + hi * 8], &sW[ct * 32 + hi * 8 + 16]);
      acc[t] = __builtin_amdgcn_wmma_f32_16x16x32_bf16(
          false, af, false, bf, (short)0, acc[t], false, false);
    }
  }

  // -------- Epilogue: out_h = h + (Y + b2), in place over agg --------
  const int Rw = R0 + waveR * 16;
#pragma unroll
  for (int t = 0; t < 8; ++t) {
    int c = waveC * 128 + t * 16 + lanelo;
    float bb = b2[c];
    int rbase = Rw + hi * 8;
#pragma unroll
    for (int j = 0; j < 8; ++j) {
      int rg = rbase + j;
      if (rg < Nn) {
        float y = acc[t][j] + bb;
        agg_out[rg * 256 + c] = h[rg * 256 + c] + y;
      }
    }
  }
}

extern "C" void kernel_launch(void* const* d_in, const int* in_sizes, int n_in,
                              void* d_out, int out_size, void* d_ws, size_t ws_size,
                              hipStream_t stream) {
  (void)n_in; (void)out_size; (void)ws_size;
  const float* h    = (const float*)d_in[0];
  const float* e    = (const float*)d_in[1];
  const float* W_e1 = (const float*)d_in[2];
  const float* b_e1 = (const float*)d_in[3];
  const float* W_e2 = (const float*)d_in[4];
  // d_in[5] = b_e2 (cancels in mf, unused)
  const float* W_n1 = (const float*)d_in[6];
  const float* b_n1 = (const float*)d_in[7];
  const float* W_n2 = (const float*)d_in[8];
  const float* b_n2 = (const float*)d_in[9];
  const int*   ei   = (const int*)d_in[10];

  const int H  = 256;
  const int Nn = in_sizes[0] / H;    // 20000
  const int E  = in_sizes[10] / 2;   // 320000

  float* out_h = (float*)d_out;                  // doubles as agg scratch
  float* e2    = out_h + (size_t)Nn * H;

  // Staged bf16 weights in workspace (~0.9 MB): 56 k-steps * 8192 elements.
  __bf16* Wt    = (__bf16*)d_ws;
  __bf16* Wt_e1 = Wt;                 // 24 k-steps
  __bf16* Wt_e2 = Wt + 24 * 8192;    //  8 k-steps
  __bf16* Wt_n1 = Wt + 32 * 8192;    // 16 k-steps
  __bf16* Wt_n2 = Wt + 48 * 8192;    //  8 k-steps

  const int totalW = 56 * 8192;
  convert_weights_kernel<<<(totalW + 255) / 256, 256, 0, stream>>>(
      W_e1, W_e2, W_n1, W_n2, Wt);

  // agg = 0 (graph-capture-safe stream memset)
  (void)hipMemsetAsync(out_h, 0, (size_t)Nn * H * sizeof(float), stream);

  const int blocksE = (E + 63) / 64;
  edge_mlp_kernel<<<blocksE, 256, 0, stream>>>(
      h, e, Wt_e1, b_e1, Wt_e2, ei, ei + E, e2, out_h, E);

  const int blocksN = (Nn + 63) / 64;
  node_mlp_kernel<<<blocksN, 256, 0, stream>>>(
      h, Wt_n1, b_n1, Wt_n2, b_n2, out_h, Nn);
}
